// CausalWanSelfAttention_59863254172115
// MI455X (gfx1250) — matrix operand order repaired and checked by reading, not compiled, not run on hardware
//
#include <hip/hip_runtime.h>

typedef __bf16 bf16;
typedef __attribute__((ext_vector_type(16))) __bf16 v16bf;
typedef __attribute__((ext_vector_type(8)))  __bf16 v8bf;
typedef __attribute__((ext_vector_type(8)))  float  v8f;
typedef __attribute__((__vector_size__(16))) int    v4i;

#define DIM 1536
#define NH 12
#define HD 128
#define SQ 3120
#define CACHE_LEN 6240

#define AS1 __attribute__((address_space(1)))
#define AS3 __attribute__((address_space(3)))

#if __has_builtin(__builtin_amdgcn_global_load_async_to_lds_b128)
#define ASYNC_B128(g, l, off) \
  __builtin_amdgcn_global_load_async_to_lds_b128((AS1 v4i*)(void*)(g), (AS3 v4i*)(void*)(l), (off), 0)
#define WAIT_ASYNC(n) asm volatile("s_wait_asynccnt " #n ::: "memory")
#else
#define ASYNC_B128(g, l, off) \
  do { *(v8bf*)((char*)(void*)(l) + (off)) = *(const v8bf*)((const char*)(const void*)(g) + (off)); } while (0)
#define WAIT_ASYNC(n) do {} while (0)
#endif

static __device__ __forceinline__ unsigned short f2bfu(float f) {
  union { float f; unsigned u; } c; c.f = f;
  unsigned u = c.u + 0x7FFFu + ((c.u >> 16) & 1u);
  return (unsigned short)(u >> 16);
}
static __device__ __forceinline__ bf16 f2bf(float f) {
  union { unsigned short s; bf16 b; } o; o.s = f2bfu(f);
  return o.b;
}
static __device__ __forceinline__ v8f vzero() {
  v8f z;
#pragma unroll
  for (int i = 0; i < 8; ++i) z[i] = 0.f;
  return z;
}

// A fragment, 16x32 bf16, row-major source with leading dim ld.
static __device__ __forceinline__ v16bf load_a_frag(const bf16* base, int ld, int k0, int lane) {
  int half = lane >> 4, r = lane & 15;
  const bf16* p = base + (size_t)r * ld + k0 + half * 16;
  v8bf a0 = *(const v8bf*)p;
  v8bf a1 = *(const v8bf*)(p + 8);
  return __builtin_shufflevector(a0, a1, 0,1,2,3,4,5,6,7,8,9,10,11,12,13,14,15);
}
// B fragment from B-transposed storage Bt[N][K].
static __device__ __forceinline__ v16bf load_bt_frag(const bf16* base, int ld, int k0, int lane) {
  int half = lane >> 4, r = lane & 15;
  return *(const v16bf*)(base + (size_t)r * ld + k0 + half * 16);
}
static __device__ __forceinline__ v8f wmma_bf16(v16bf a, v16bf b, v8f c) {
  return __builtin_amdgcn_wmma_f32_16x16x32_bf16(false, a, false, b, (short)0, c, false, false);
}

// ---------------- casts ----------------
__global__ void k_cast(const float* __restrict__ s, bf16* __restrict__ d, int n) {
  int i = blockIdx.x * 256 + threadIdx.x;
  if (i < n) d[i] = f2bf(s[i]);
}
__global__ void k_castw(const float* __restrict__ W, bf16* __restrict__ Wt) {
  int i = blockIdx.x * 256 + threadIdx.x; // i = n*DIM + k
  int n = i / DIM, k = i - n * DIM;
  Wt[i] = f2bf(W[(size_t)k * DIM + n]);
}
__global__ void k_cache_old(const float* __restrict__ kc, const float* __restrict__ vc,
                            bf16* __restrict__ kws, bf16* __restrict__ vtws) {
  int i = blockIdx.x * 256 + threadIdx.x; // over CACHE_LEN*NH*HD
  int key = i / (NH * HD);
  int rest = i - key * (NH * HD);
  int h = rest / HD, d = rest - h * HD;
  kws[((size_t)h * CACHE_LEN + key) * HD + d] = f2bf(kc[i]);
  vtws[(size_t)h * HD * CACHE_LEN + (size_t)d * CACHE_LEN + key] = f2bf(vc[i]);
}

// ---------------- GEMM: C[M][N] = A[M][K] * Bt[N][K]^T + bias ----------------
// block = 128 (4 waves). Waves share B chunks staged via async LDS copies,
// each wave owns one 16-row M tile; 64 N columns per block.
__global__ void k_gemm(const bf16* __restrict__ A, const bf16* __restrict__ Bt,
                       const float* __restrict__ bias, float* __restrict__ C,
                       int M, int N, int K) {
  __shared__ __align__(32) bf16 Bbuf[2][64 * 32];
  int tid = threadIdx.x;
  int wave = tid >> 5, lane = tid & 31;
  int half = lane >> 4, r = lane & 15;
  int mt = blockIdx.y * 4 + wave;
  int mmax = M / 16 - 1;
  if (mt > mmax) mt = mmax;           // keep waves alive for barriers
  int m0 = mt * 16, n0 = blockIdx.x * 64;
  const bf16* Arow = A + (size_t)m0 * K;
  int brow = tid >> 1, bpart = tid & 1;
  const bf16* Bg = Bt + (size_t)(n0 + brow) * K + bpart * 16;

  { bf16* lb = Bbuf[0] + brow * 32 + bpart * 16;
    ASYNC_B128(Bg, lb, 0); ASYNC_B128(Bg, lb, 16); }

  v8f acc[4];
#pragma unroll
  for (int t = 0; t < 4; ++t) acc[t] = vzero();

  const int NIT = K / 32;
  for (int it = 0; it < NIT; ++it) {
    int cur = it & 1;
    if (it + 1 < NIT) {
      bf16* lb = Bbuf[1 - cur] + brow * 32 + bpart * 16;
      const bf16* g = Bg + (size_t)(it + 1) * 32;
      ASYNC_B128(g, lb, 0); ASYNC_B128(g, lb, 16);
      WAIT_ASYNC(0x2);
    } else {
      WAIT_ASYNC(0x0);
    }
    __syncthreads();
    v16bf af = load_a_frag(Arow, K, it * 32, lane);
    const bf16* Bb = Bbuf[cur];
#pragma unroll
    for (int t = 0; t < 4; ++t) {
      v16bf bf_ = load_bt_frag(Bb + (size_t)(16 * t) * 32, 32, 0, lane);
      acc[t] = wmma_bf16(af, bf_, acc[t]);
    }
    __syncthreads();
  }
#pragma unroll
  for (int t = 0; t < 4; ++t)
#pragma unroll
    for (int i = 0; i < 8; ++i) {
      int row = m0 + i + 8 * half;
      int col = n0 + 16 * t + r;
      C[(size_t)row * N + col] = acc[t][i] + bias[col];
    }
}

// ---------------- fused RMSNorm + RoPE + cache/layout build ----------------
__global__ void k_normrope(const float* __restrict__ qf, const float* __restrict__ kf,
                           const float* __restrict__ vf,
                           const float* __restrict__ gq, const float* __restrict__ gk,
                           const float* __restrict__ freqs,
                           const int* p_gh, const int* p_gw, const int* p_kvs, const int* p_cs,
                           bf16* __restrict__ qws, bf16* __restrict__ kws, bf16* __restrict__ vtws,
                           float* __restrict__ out_k, float* __restrict__ out_v) {
  __shared__ float sq[256], sk[256];
  int row = blockIdx.x, t = threadIdx.x;
  const float* qr = qf + (size_t)row * DIM;
  const float* kr = kf + (size_t)row * DIM;
  const float* vr = vf + (size_t)row * DIM;
  float aq = 0.f, ak = 0.f;
  for (int i = t; i < DIM; i += 256) { float a = qr[i]; aq += a * a; float b = kr[i]; ak += b * b; }
  sq[t] = aq; sk[t] = ak;
  __syncthreads();
  for (int s = 128; s > 0; s >>= 1) {
    if (t < s) { sq[t] += sq[t + s]; sk[t] += sk[t + s]; }
    __syncthreads();
  }
  float rq = rsqrtf(sq[0] / DIM + 1e-6f);
  float rk = rsqrtf(sk[0] / DIM + 1e-6f);
  int gh = *p_gh, gw = *p_gw, kvs = *p_kvs, cs = *p_cs;
  int ghw = gh * gw;
  int sf = cs / ghw;
  int fi = row / ghw, rem = row - fi * ghw, hi = rem / gw, wi = rem - hi * gw;
  const int cc = HD / 2, c2 = cc / 3, c1 = cc - 2 * c2;
  const float qscale = 0.088388347648318447f; // 1/sqrt(HD), folded into Q
  for (int p = t; p < DIM / 2; p += 256) {
    int h = p >> 6, jj = p & 63;
    int col = 2 * p;
    int frow = (jj < c1) ? (sf + fi) : ((jj < c1 + c2) ? hi : wi);
    float ang = freqs[frow * cc + jj];
    float sn, csn; __sincosf(ang, &sn, &csn);
    float xr = qr[col] * rq * gq[col];
    float xi = qr[col + 1] * rq * gq[col + 1];
    float orr = xr * csn - xi * sn, oi = xr * sn + xi * csn;
    size_t qb = ((size_t)h * SQ + row) * HD + 2 * jj;
    qws[qb] = f2bf(orr * qscale); qws[qb + 1] = f2bf(oi * qscale);
    xr = kr[col] * rk * gk[col];
    xi = kr[col + 1] * rk * gk[col + 1];
    orr = xr * csn - xi * sn; oi = xr * sn + xi * csn;
    out_k[(size_t)row * DIM + col] = orr;
    out_k[(size_t)row * DIM + col + 1] = oi;
    size_t kb = ((size_t)h * CACHE_LEN + (kvs + row)) * HD + 2 * jj;
    kws[kb] = f2bf(orr); kws[kb + 1] = f2bf(oi);
  }
  for (int i = t; i < DIM; i += 256) {
    float v = vr[i];
    out_v[(size_t)row * DIM + i] = v;
    int h = i >> 7, d = i & 127;
    vtws[(size_t)h * HD * CACHE_LEN + (size_t)d * CACHE_LEN + (kvs + row)] = f2bf(v);
  }
}

// ---------------- flash attention ----------------
// block 128 (4 waves); all waves share one head and the K/V chunk stream,
// staged via double-buffered async LDS copies. Each wave owns 16 query rows.
__global__ void k_flash(const bf16* __restrict__ Q, const bf16* __restrict__ Kc,
                        const bf16* __restrict__ Vt, float* __restrict__ O) {
  __shared__ __align__(32) bf16 Kbuf[2][32 * HD];   // [key][d]
  __shared__ __align__(32) bf16 Vbuf[2][HD * 32];   // [d][key]
  __shared__ __align__(32) unsigned short psh[4][16 * 32];
  int tid = threadIdx.x;
  int head = blockIdx.x;
  int wave = tid >> 5, lane = tid & 31;
  int half = lane >> 4, r = lane & 15;
  int qtile = blockIdx.y * 4 + wave;
  if (qtile > SQ / 16 - 1) qtile = SQ / 16 - 1;     // keep waves alive for barriers
  int qbase = qtile * 16;
  const bf16* Qh = Q + ((size_t)head * SQ + qbase) * HD;
  const bf16* Kh = Kc + (size_t)head * CACHE_LEN * HD;
  const bf16* Vh = Vt + (size_t)head * HD * CACHE_LEN;
  v16bf qa[4];
#pragma unroll
  for (int dc = 0; dc < 4; ++dc) qa[dc] = load_a_frag(Qh, HD, dc * 32, lane);
  float m[8], l[8]; v8f o[8];
#pragma unroll
  for (int i = 0; i < 8; ++i) { m[i] = -3.0e38f; l[i] = 0.f; o[i] = vzero(); }
  unsigned short* ps = psh[wave];

  int kk = tid >> 2, kp = tid & 3;  // K chunk: row kk (of 32), 64B quarter kp

  { const bf16* g = Kh + (size_t)kk * HD + kp * 32;
    bf16* lk = Kbuf[0] + kk * HD + kp * 32;
    ASYNC_B128(g, lk, 0); ASYNC_B128(g, lk, 16); ASYNC_B128(g, lk, 32); ASYNC_B128(g, lk, 48);
    const bf16* gv = Vh + (size_t)tid * CACHE_LEN;
    bf16* lv = Vbuf[0] + tid * 32;
    ASYNC_B128(gv, lv, 0); ASYNC_B128(gv, lv, 16); ASYNC_B128(gv, lv, 32); ASYNC_B128(gv, lv, 48);
  }

  const int NIT = CACHE_LEN / 32; // 195
  for (int it = 0; it < NIT; ++it) {
    int cur = it & 1;
    if (it + 1 < NIT) {
      int kb2 = (it + 1) * 32;
      const bf16* g = Kh + (size_t)(kb2 + kk) * HD + kp * 32;
      bf16* lk = Kbuf[1 - cur] + kk * HD + kp * 32;
      ASYNC_B128(g, lk, 0); ASYNC_B128(g, lk, 16); ASYNC_B128(g, lk, 32); ASYNC_B128(g, lk, 48);
      const bf16* gv = Vh + (size_t)tid * CACHE_LEN + kb2;
      bf16* lv = Vbuf[1 - cur] + tid * 32;
      ASYNC_B128(gv, lv, 0); ASYNC_B128(gv, lv, 16); ASYNC_B128(gv, lv, 32); ASYNC_B128(gv, lv, 48);
      WAIT_ASYNC(0x8);
    } else {
      WAIT_ASYNC(0x0);
    }
    __syncthreads();
    const bf16* Kb = Kbuf[cur];
    const bf16* Vb = Vbuf[cur];
    // S = Q (16x128) * K_chunk^T -> two 16x16 f32 tiles
    v8f s0 = vzero(), s1 = vzero();
#pragma unroll
    for (int dc = 0; dc < 4; ++dc) {
      v16bf b0 = load_bt_frag(Kb, HD, dc * 32, lane);
      v16bf b1 = load_bt_frag(Kb + 16 * HD, HD, dc * 32, lane);
      s0 = wmma_bf16(qa[dc], b0, s0);
      s1 = wmma_bf16(qa[dc], b1, s1);
    }
    // online softmax (row stats via shfl_xor within 16-lane halves)
    float alpha[8];
#pragma unroll
    for (int i = 0; i < 8; ++i) {
      float v = fmaxf(s0[i], s1[i]);
#pragma unroll
      for (int off = 1; off <= 8; off <<= 1) v = fmaxf(v, __shfl_xor(v, off, 32));
      float mn = fmaxf(m[i], v);
      float al = __expf(m[i] - mn);
      float p0 = __expf(s0[i] - mn);
      float p1 = __expf(s1[i] - mn);
      float rs = p0 + p1;
#pragma unroll
      for (int off = 1; off <= 8; off <<= 1) rs += __shfl_xor(rs, off, 32);
      l[i] = l[i] * al + rs; m[i] = mn; alpha[i] = al;
      s0[i] = p0; s1[i] = p1;
    }
#pragma unroll
    for (int j = 0; j < 8; ++j)
#pragma unroll
      for (int i = 0; i < 8; ++i) o[j][i] *= alpha[i];
    // P: C-layout -> LDS -> A-layout
#pragma unroll
    for (int i = 0; i < 8; ++i) {
      int rowi = i + 8 * half;
      ps[rowi * 32 + r]      = f2bfu(s0[i]);
      ps[rowi * 32 + 16 + r] = f2bfu(s1[i]);
    }
    asm volatile("s_wait_dscnt 0" ::: "memory");
    v16bf pa = load_a_frag((const bf16*)ps, 32, 0, lane);
#pragma unroll
    for (int j = 0; j < 8; ++j) {
      v16bf bv = load_bt_frag(Vb + (size_t)(j * 16) * 32, 32, 0, lane);
      o[j] = wmma_bf16(pa, bv, o[j]);
    }
    __syncthreads();
  }
#pragma unroll
  for (int i = 0; i < 8; ++i) l[i] = 1.0f / l[i];
#pragma unroll
  for (int j = 0; j < 8; ++j)
#pragma unroll
    for (int i = 0; i < 8; ++i) {
      int row = qbase + i + 8 * half;
      int col = head * HD + j * 16 + r;
      O[(size_t)row * DIM + col] = o[j][i] * l[i];
    }
}

extern "C" void kernel_launch(void* const* d_in, const int* in_sizes, int n_in,
                              void* d_out, int out_size, void* d_ws, size_t ws_size,
                              hipStream_t stream) {
  (void)in_sizes; (void)n_in; (void)out_size; (void)ws_size;
  const float* x      = (const float*)d_in[0];
  const float* freqs  = (const float*)d_in[1];
  const float* Wq     = (const float*)d_in[2];
  const float* bq     = (const float*)d_in[3];
  const float* Wk     = (const float*)d_in[4];
  const float* bk     = (const float*)d_in[5];
  const float* Wv     = (const float*)d_in[6];
  const float* bv     = (const float*)d_in[7];
  const float* Wo     = (const float*)d_in[8];
  const float* bo     = (const float*)d_in[9];
  const float* gq     = (const float*)d_in[10];
  const float* gk     = (const float*)d_in[11];
  const float* kcache = (const float*)d_in[12];
  const float* vcache = (const float*)d_in[13];
  const int*   p_gh   = (const int*)d_in[15];
  const int*   p_gw   = (const int*)d_in[16];
  const int*   p_kvs  = (const int*)d_in[17];
  const int*   p_cs   = (const int*)d_in[19];

  char* ws = (char*)d_ws;
  size_t cur = 0;
  auto take = [&](size_t bytes) -> char* {
    cur = (cur + 255) & ~(size_t)255;
    char* p = ws + cur; cur += bytes; return p;
  };
  bf16* xbf  = (bf16*)take((size_t)SQ * DIM * 2);
  bf16* wqt  = (bf16*)take((size_t)DIM * DIM * 2);
  bf16* wkt  = (bf16*)take((size_t)DIM * DIM * 2);
  bf16* wvt  = (bf16*)take((size_t)DIM * DIM * 2);
  bf16* wot  = (bf16*)take((size_t)DIM * DIM * 2);
  float* qf  = (float*)take((size_t)SQ * DIM * 4);
  float* kf  = (float*)take((size_t)SQ * DIM * 4);
  float* vf  = (float*)take((size_t)SQ * DIM * 4);
  bf16* qws  = (bf16*)take((size_t)NH * SQ * HD * 2);
  bf16* kws  = (bf16*)take((size_t)NH * CACHE_LEN * HD * 2);
  bf16* vtws = (bf16*)take((size_t)NH * HD * CACHE_LEN * 2);
  float* ao  = kf;   // reuse: kf is consumed by k_normrope before k_flash writes ao
  bf16* aobf = xbf;  // reuse: xbf is consumed by the QKV GEMMs before the cast

  float* out0  = (float*)d_out;
  float* out_k = out0 + (size_t)SQ * DIM;
  float* out_v = out_k + (size_t)SQ * DIM;

  const int nW = DIM * DIM;
  k_castw<<<nW / 256, 256, 0, stream>>>(Wq, wqt);
  k_castw<<<nW / 256, 256, 0, stream>>>(Wk, wkt);
  k_castw<<<nW / 256, 256, 0, stream>>>(Wv, wvt);
  k_castw<<<nW / 256, 256, 0, stream>>>(Wo, wot);
  k_cast<<<(SQ * DIM) / 256, 256, 0, stream>>>(x, xbf, SQ * DIM);

  dim3 gg(DIM / 64, (SQ / 16 + 3) / 4);
  k_gemm<<<gg, 128, 0, stream>>>(xbf, wqt, bq, qf, SQ, DIM, DIM);
  k_gemm<<<gg, 128, 0, stream>>>(xbf, wkt, bk, kf, SQ, DIM, DIM);
  k_gemm<<<gg, 128, 0, stream>>>(xbf, wvt, bv, vf, SQ, DIM, DIM);

  k_cache_old<<<(CACHE_LEN * NH * HD) / 256, 256, 0, stream>>>(kcache, vcache, kws, vtws);
  k_normrope<<<SQ, 256, 0, stream>>>(qf, kf, vf, gq, gk, freqs, p_gh, p_gw, p_kvs, p_cs,
                                     qws, kws, vtws, out_k, out_v);

  dim3 gf(NH, (SQ / 16 + 3) / 4);
  k_flash<<<gf, 128, 0, stream>>>(qws, kws, vtws, ao);

  k_cast<<<(SQ * DIM) / 256, 256, 0, stream>>>(ao, aobf, SQ * DIM);
  k_gemm<<<gg, 128, 0, stream>>>(aobf, wot, bo, out0, SQ, DIM, DIM);
}
